// StructuralTypeSeqModel_55164559949892
// MI455X (gfx1250) — compile-verified, matrix operand order checked by reading, not run on hardware
//
#include <hip/hip_runtime.h>
#include <hip/hip_bf16.h>

typedef __attribute__((ext_vector_type(16))) __bf16 v16bf;
typedef __attribute__((ext_vector_type(8)))  __bf16 v8bf;
typedef __attribute__((ext_vector_type(8)))  float  v8f;

#define H_DIM 64
#define NEG_SLOPE 0.2f
#define GAT_EPS 1e-16f

// ---------------- elementwise helpers ----------------

__global__ void k_f32_to_bf16(const float* __restrict__ in, __bf16* __restrict__ out, int n) {
  int i = blockIdx.x * blockDim.x + threadIdx.x;
  if (i < n) out[i] = (__bf16)in[i];
}

// Pack W [K x 64] fp32 row-major into the per-lane WMMA B-operand layout:
// index = ((kb*4 + nb)*32 + lane)*16 + j ; lane: n=lane&15, half=lane>>4 ;
// j<8  -> K = half*8 + j ; j>=8 -> K = 16 + half*8 + (j-8)   (ISA 7.12.2, 16-bit B)
__global__ void k_pack_w(const float* __restrict__ W, __bf16* __restrict__ Wp, int kb_count) {
  int tid = blockIdx.x * blockDim.x + threadIdx.x;
  int total = kb_count * 4 * 512;
  if (tid >= total) return;
  int tile = tid >> 9;
  int r    = tid & 511;
  int lane = r >> 4;
  int j    = r & 15;
  int kb = tile >> 2;
  int nb = tile & 3;
  int n    = lane & 15;
  int half = lane >> 4;
  int kk = (j < 8) ? (half * 8 + j) : (16 + half * 8 + (j - 8));
  Wp[tid] = (__bf16)W[(size_t)(kb * 32 + kk) * H_DIM + nb * 16 + n];
}

// ---------------- WMMA GEMM: H[N x 64] = A[N x K](bf16) * W(packed bf16) ----------------
// One wave per 16x16 output tile; 4 waves per block cover the 64 output columns.
__global__ void k_gemm_wmma(const __bf16* __restrict__ A, const __bf16* __restrict__ Wp,
                            float* __restrict__ Hout, int Nrows, int K) {
  int wave = threadIdx.x >> 5;   // n-tile (0..3)
  int lane = threadIdx.x & 31;
  int tile_m = blockIdx.x;
  int n    = lane & 15;
  int half = lane >> 4;
  int row = tile_m * 16 + n;     // A-matrix: lanes 0-15 and 16-31 both hold rows M=0..15
  if (row >= Nrows) row = Nrows - 1;
  v8f acc = {};
  int kb_count = K >> 5;
  for (int kb = 0; kb < kb_count; ++kb) {
    int k0 = kb * 32 + half * 8;
    v8bf c0 = *(const v8bf*)(A + (size_t)row * K + k0);        // K = half*8 + 0..7
    v8bf c1 = *(const v8bf*)(A + (size_t)row * K + k0 + 16);   // K = 16 + half*8 + 0..7
    v16bf a;
#pragma unroll
    for (int j = 0; j < 8; ++j) { a[j] = c0[j]; a[8 + j] = c1[j]; }
    v16bf b = *(const v16bf*)(Wp + ((size_t)(kb * 4 + wave) * 32 + lane) * 16);
    acc = __builtin_amdgcn_wmma_f32_16x16x32_bf16(false, a, false, b, (short)0, acc,
                                                  false, false);
  }
  int mbase = tile_m * 16 + half * 8;
#pragma unroll
  for (int v = 0; v < 8; ++v) {
    int m = mbase + v;
    if (m < Nrows) Hout[(size_t)m * H_DIM + wave * 16 + n] = acc[v];
  }
}

// ---------------- attention scalar dots: a_src[i] = h[i].as , a_dst[i] = h[i].ad ----------------
__global__ void k_attdot(const float* __restrict__ Hh, const float* __restrict__ avs,
                         const float* __restrict__ avd, float* __restrict__ ASRC,
                         float* __restrict__ ADST, int N) {
  int wave = threadIdx.x >> 5;
  int lane = threadIdx.x & 31;
  int node = blockIdx.x * 8 + wave;
  if (node >= N) return;
  float h0 = Hh[(size_t)node * H_DIM + lane];
  float h1 = Hh[(size_t)node * H_DIM + lane + 32];
  float vs = h0 * avs[lane] + h1 * avs[lane + 32];
  float vd = h0 * avd[lane] + h1 * avd[lane + 32];
#pragma unroll
  for (int off = 16; off >= 1; off >>= 1) {
    vs += __shfl_xor(vs, off, 32);
    vd += __shfl_xor(vd, off, 32);
  }
  if (lane == 0) { ASRC[node] = vs; ADST[node] = vd; }
}

__global__ void k_init(float* __restrict__ ACC, float* __restrict__ Mx,
                       float* __restrict__ Ss, int N) {
  int i = blockIdx.x * blockDim.x + threadIdx.x;
  if (i < N * H_DIM) ACC[i] = 0.0f;
  if (i < N) { Mx[i] = -3.0e38f; Ss[i] = 0.0f; }
}

__device__ inline void atomicMaxF(float* addr, float value) {
  unsigned int* ai = (unsigned int*)addr;
  unsigned int old = *ai;
  while (__uint_as_float(old) < value) {
    unsigned int assumed = old;
    old = atomicCAS(ai, assumed, __float_as_uint(value));
    if (old == assumed) break;
  }
}

// pass 1: e = leaky_relu(a_src[s]+a_dst[d]); segment max over dst
__global__ void k_edge_max(const int* __restrict__ src, const int* __restrict__ dst,
                           const float* __restrict__ ASRC, const float* __restrict__ ADST,
                           float* __restrict__ EXB, float* __restrict__ Mx, int E, int Etot) {
  int e = blockIdx.x * blockDim.x + threadIdx.x;
  if (e >= Etot) return;
  int s = (e < E) ? src[e] : (e - E);
  int d = (e < E) ? dst[e] : (e - E);
  float v = ASRC[s] + ADST[d];
  v = (v > 0.0f) ? v : NEG_SLOPE * v;
  EXB[e] = v;
  atomicMaxF(&Mx[d], v);
}

// pass 2: ex = exp(e - m[d]); segment sum over dst
__global__ void k_edge_exp(const int* __restrict__ dst, float* __restrict__ EXB,
                           const float* __restrict__ Mx, float* __restrict__ Ss,
                           int E, int Etot) {
  int e = blockIdx.x * blockDim.x + threadIdx.x;
  if (e >= Etot) return;
  int d = (e < E) ? dst[e] : (e - E);
  float ex = __expf(EXB[e] - Mx[d]);
  EXB[e] = ex;
  atomicAdd(&Ss[d], ex);
}

// pass 3: ACC[d] += alpha * h[s]; one wave per edge, 2 features per lane
__global__ void k_edge_scatter(const int* __restrict__ src, const int* __restrict__ dst,
                               const float* __restrict__ EXB, const float* __restrict__ Ss,
                               const float* __restrict__ Hh, float* __restrict__ ACC,
                               int E, int Etot) {
  long long gid = (long long)blockIdx.x * blockDim.x + threadIdx.x;
  int e = (int)(gid >> 5);
  int lane = (int)(gid & 31);
  if (e >= Etot) return;
  int s = (e < E) ? src[e] : (e - E);
  int d = (e < E) ? dst[e] : (e - E);
  float alpha = EXB[e] / (Ss[d] + GAT_EPS);
  atomicAdd(&ACC[(size_t)d * H_DIM + lane],      alpha * Hh[(size_t)s * H_DIM + lane]);
  atomicAdd(&ACC[(size_t)d * H_DIM + lane + 32], alpha * Hh[(size_t)s * H_DIM + lane + 32]);
}

// bias + optional relu, optional bf16 convert for next layer's GEMM input
__global__ void k_bias_act(float* __restrict__ ACC, const float* __restrict__ bias,
                           __bf16* __restrict__ Anext, int N, int relu, int convert) {
  int i = blockIdx.x * blockDim.x + threadIdx.x;
  if (i >= N * H_DIM) return;
  float v = ACC[i] + bias[i & 63];
  if (relu) v = fmaxf(v, 0.0f);
  ACC[i] = v;
  if (convert) Anext[i] = (__bf16)v;
}

// node0[g] = lower_bound(batch, g)  (batch sorted)
__global__ void k_node0(const int* __restrict__ batch, int* __restrict__ node0, int N, int B) {
  int b = blockIdx.x * blockDim.x + threadIdx.x;
  if (b >= B) return;
  int lo = 0, hi = N;
  while (lo < hi) { int mid = (lo + hi) >> 1; if (batch[mid] < b) lo = mid + 1; else hi = mid; }
  node0[b] = lo;
}

// logits[b,c] = h3[node0[b]] . Wp[:,c] + bp[c]
__global__ void k_proj(const float* __restrict__ H3, const int* __restrict__ node0,
                       const float* __restrict__ Wp, const float* __restrict__ bp,
                       float* __restrict__ out, int B, int C) {
  int t = blockIdx.x * blockDim.x + threadIdx.x;
  if (t >= B * C) return;
  int b = t / C, c = t % C;
  const float* h = H3 + (size_t)node0[b] * H_DIM;
  float acc = bp[c];
#pragma unroll 8
  for (int k = 0; k < H_DIM; ++k) acc += h[k] * Wp[k * C + c];
  out[t] = acc;
}

// ---------------- host driver ----------------

extern "C" void kernel_launch(void* const* d_in, const int* in_sizes, int n_in,
                              void* d_out, int out_size, void* d_ws, size_t ws_size,
                              hipStream_t stream) {
  (void)n_in; (void)ws_size;
  const float* x     = (const float*)d_in[0];
  const int*   eidx  = (const int*)d_in[1];
  const int*   batch = (const int*)d_in[2];
  const float* W1 = (const float*)d_in[4];
  const float* as1 = (const float*)d_in[5];
  const float* ad1 = (const float*)d_in[6];
  const float* b1 = (const float*)d_in[7];
  const float* W2 = (const float*)d_in[8];
  const float* as2 = (const float*)d_in[9];
  const float* ad2 = (const float*)d_in[10];
  const float* b2 = (const float*)d_in[11];
  const float* W3 = (const float*)d_in[12];
  const float* as3 = (const float*)d_in[13];
  const float* ad3 = (const float*)d_in[14];
  const float* b3 = (const float*)d_in[15];
  const float* Wp = (const float*)d_in[16];
  const float* bp = (const float*)d_in[17];

  const int N    = in_sizes[2];            // batch vector length = #nodes
  const int F_IN = in_sizes[0] / N;        // 128
  const int E    = in_sizes[1] / 2;        // 3.2M
  const int C    = in_sizes[16] / H_DIM;   // 32
  const int B    = out_size / C;           // 1000
  const int Etot = E + N;
  const int* srcA = eidx;
  const int* dstA = eidx + E;

  size_t off = 0; char* base = (char*)d_ws;
  auto alloc = [&](size_t bytes) -> void* {
    off = (off + 255) & ~(size_t)255;
    void* p = base + off; off += bytes; return p;
  };
  __bf16* Abf  = (__bf16*)alloc((size_t)N * F_IN * sizeof(__bf16));
  float*  Hh   = (float*)alloc((size_t)N * H_DIM * sizeof(float));
  float*  ACC  = (float*)alloc((size_t)N * H_DIM * sizeof(float));
  float*  ASRC = (float*)alloc((size_t)N * sizeof(float));
  float*  ADST = (float*)alloc((size_t)N * sizeof(float));
  float*  Mx   = (float*)alloc((size_t)N * sizeof(float));
  float*  Ss   = (float*)alloc((size_t)N * sizeof(float));
  float*  EXB  = (float*)alloc((size_t)Etot * sizeof(float));
  __bf16* WP   = (__bf16*)alloc((size_t)(F_IN / 32) * 4 * 512 * sizeof(__bf16));
  int*    ND0  = (int*)alloc((size_t)B * sizeof(int));

  auto cdiv = [](long long a, long long b) { return (int)((a + b - 1) / b); };

  auto layer = [&](const __bf16* Ain, int K, const float* W, const float* avs,
                   const float* avd, const float* bias, int relu, int convert) {
    int kb = K / 32;
    k_pack_w<<<cdiv((long long)kb * 4 * 512, 256), 256, 0, stream>>>(W, WP, kb);
    k_gemm_wmma<<<cdiv(N, 16), 128, 0, stream>>>(Ain, WP, Hh, N, K);
    k_attdot<<<cdiv(N, 8), 256, 0, stream>>>(Hh, avs, avd, ASRC, ADST, N);
    k_init<<<cdiv((long long)N * H_DIM, 256), 256, 0, stream>>>(ACC, Mx, Ss, N);
    k_edge_max<<<cdiv(Etot, 256), 256, 0, stream>>>(srcA, dstA, ASRC, ADST, EXB, Mx, E, Etot);
    k_edge_exp<<<cdiv(Etot, 256), 256, 0, stream>>>(dstA, EXB, Mx, Ss, E, Etot);
    k_edge_scatter<<<cdiv((long long)Etot * 32, 256), 256, 0, stream>>>(
        srcA, dstA, EXB, Ss, Hh, ACC, E, Etot);
    k_bias_act<<<cdiv((long long)N * H_DIM, 256), 256, 0, stream>>>(
        ACC, bias, Abf, N, relu, convert);
  };

  k_f32_to_bf16<<<cdiv((long long)N * F_IN, 256), 256, 0, stream>>>(x, Abf, N * F_IN);
  layer(Abf, F_IN,  W1, as1, ad1, b1, /*relu=*/1, /*convert=*/1);
  layer(Abf, H_DIM, W2, as2, ad2, b2, /*relu=*/1, /*convert=*/1);
  layer(Abf, H_DIM, W3, as3, ad3, b3, /*relu=*/0, /*convert=*/0);

  k_node0<<<cdiv(B, 256), 256, 0, stream>>>(batch, ND0, N, B);
  k_proj<<<cdiv((long long)B * C, 256), 256, 0, stream>>>(ACC, ND0, Wp, bp,
                                                          (float*)d_out, B, C);
}
